// Attention_71090298683696
// MI455X (gfx1250) — compile-verified, matrix operand order checked by reading, not run on hardware
//
#include <hip/hip_runtime.h>
#include <hip/hip_bf16.h>
#include <stdint.h>

typedef __attribute__((ext_vector_type(16))) __bf16 v16bf;
typedef __attribute__((ext_vector_type(8)))  float  v8f;
typedef unsigned int uint32x4 __attribute__((ext_vector_type(4)));
typedef int          int32x4  __attribute__((ext_vector_type(4)));
typedef int          int32x8  __attribute__((ext_vector_type(8)));

#if defined(__has_builtin)
#if __has_builtin(__builtin_amdgcn_tensor_load_to_lds)
#define HAVE_TDM 1
#endif
#endif

union FragBF { v16bf v; uint4 q[2]; };
union Pack8  { uint4 q; __bf16 b[8]; };

__device__ __forceinline__ __bf16 f2bf(float f) {
    union { float f; uint32_t u; } a; a.f = f;
    uint32_t r = a.u + 0x7FFFu + ((a.u >> 16) & 1u);   // round-to-nearest-even
    union { uint16_t s; __bf16 b; } o; o.s = (uint16_t)(r >> 16);
    return o.b;
}

// Load a 16x32 bf16 WMMA operand fragment from row-major LDS (row stride `ld`
// elements). Per CDNA5 ISA: lane L<16 -> row L, K = {0..7, 16..23};
// lane L>=16 -> row L-16, K = {8..15, 24..31}. Two 16B LDS reads per lane.
__device__ __forceinline__ v16bf ld_frag(const __bf16* p0, int ld) {
    int lane = threadIdx.x & 31;
    const __bf16* p = p0 + (lane & 15) * ld + ((lane & 16) ? 8 : 0);
    FragBF u;
    u.q[0] = *(const uint4*)(p);        // K khalf .. khalf+7
    u.q[1] = *(const uint4*)(p + 16);   // K khalf+16 .. khalf+23
    return u.v;
}

#ifdef HAVE_TDM
// Issue one TDM 2D tile load (rows x row_elems bf16, row stride stride_elems)
// from global memory into LDS at byte offset lds_off. D# packing per
// CDNA5 ISA (group0: count/lds/global/type, group1:
// data_size + tensor/tile dims + strides). Groups 2/3 zero (<=2D tensor).
__device__ __forceinline__ void tdm_load_2d(uint32_t lds_off, const void* gptr,
                                            uint32_t rows, uint32_t row_elems,
                                            uint32_t stride_elems) {
    uint64_t ga = (uint64_t)(uintptr_t)gptr;
    uint32_t ga_lo = (uint32_t)__builtin_amdgcn_readfirstlane((int)(uint32_t)ga);
    uint32_t ga_hi = (uint32_t)__builtin_amdgcn_readfirstlane((int)(uint32_t)(ga >> 32));
    uint32_t ldso  = (uint32_t)__builtin_amdgcn_readfirstlane((int)lds_off);

    uint32x4 g0;
    g0[0] = 1u;                                   // count=1, user descriptor
    g0[1] = ldso;                                 // lds_addr (bytes)
    g0[2] = ga_lo;                                // global_addr[31:0]
    g0[3] = (ga_hi & 0x01FFFFFFu) | (2u << 30);   // global_addr[56:32] | type=2

    int32x8 g1;
    g1[0] = (int)(1u << 16);                              // data_size = 2 bytes
    g1[1] = (int)((row_elems & 0xFFFFu) << 16);           // tensor_dim0[15:0]
    g1[2] = (int)((row_elems >> 16) |
                  ((0x100000u & 0xFFFFu) << 16));         // dim0 hi | tensor_dim1 lo
    g1[3] = (int)((0x100000u >> 16) |                     // tensor_dim1 hi (=1<<20)
                  ((row_elems & 0xFFFFu) << 16));         // tile_dim0
    g1[4] = (int)(rows & 0xFFFFu);                        // tile_dim1 (tile_dim2=0)
    g1[5] = (int)stride_elems;                            // tensor_dim0_stride lo
    g1[6] = 0;                                            // stride0 hi | stride1 lo
    g1[7] = 0;                                            // stride1 hi

    int32x4 gz = {0, 0, 0, 0};
#if __clang_major__ >= 23
    int32x8 gz8 = {0, 0, 0, 0, 0, 0, 0, 0};
    __builtin_amdgcn_tensor_load_to_lds(g0, g1, gz, gz, gz8, 0);
#else
    __builtin_amdgcn_tensor_load_to_lds(g0, g1, gz, gz, 0);
#endif
}
#endif

// ---------------------------------------------------------------- converters
__global__ __launch_bounds__(256)
void cvt_bf16_kernel(const float* __restrict__ in, __bf16* __restrict__ out, int n) {
    int i = (blockIdx.x * 256 + threadIdx.x) * 8;
    if (i + 8 <= n) {
        float4 f0 = *(const float4*)(in + i);
        float4 f1 = *(const float4*)(in + i + 4);
        Pack8 p;
        p.b[0] = f2bf(f0.x); p.b[1] = f2bf(f0.y); p.b[2] = f2bf(f0.z); p.b[3] = f2bf(f0.w);
        p.b[4] = f2bf(f1.x); p.b[5] = f2bf(f1.y); p.b[6] = f2bf(f1.z); p.b[7] = f2bf(f1.w);
        *(uint4*)(out + i) = p.q;
    }
}

// W [K,N] f32 row-major  ->  WT [N,K] bf16 row-major (transposed)
__global__ __launch_bounds__(256)
void cvt_transpose_kernel(const float* __restrict__ W, __bf16* __restrict__ WT,
                          int K, int N) {
    int idx = blockIdx.x * 256 + threadIdx.x;
    if (idx < K * N) {
        int k = idx / N, n = idx % N;
        WT[(size_t)n * K + k] = f2bf(W[idx]);
    }
}

// ------------------------------------------------------------------- GEMM
// C[M,N] = A[M,K](bf16,row-major) x BT[N,K](bf16,row-major,=B^T) + bias
// Block tile 128x128, 8 waves (wm in {0,1} x wn in {0..3}), wave tile 64x32.
// All 6 fragments for a K-step are preloaded before the 8-WMMA burst so the
// LDS waits amortize instead of gating every WMMA.
template <bool BF16OUT>
__global__ __launch_bounds__(256)
void gemm_bias_kernel(const __bf16* __restrict__ A, const __bf16* __restrict__ BT,
                      const float* __restrict__ bias, void* __restrict__ Cout,
                      int M, int N, int K) {
    __shared__ __align__(16) __bf16 As[128 * 32];
    __shared__ __align__(16) __bf16 Bs[128 * 32];
    int tid = threadIdx.x;
    int m0 = blockIdx.x * 128, n0 = blockIdx.y * 128;
    int w = tid >> 5, wm = w & 1, wn = w >> 1;
    int lane = tid & 31;

    v8f acc[8];
    for (int t = 0; t < 8; ++t)
        for (int e = 0; e < 8; ++e) acc[t][e] = 0.0f;

    int lrow = tid >> 1, lcol = (tid & 1) * 16;
    const __bf16* gA = A  + (size_t)(m0 + lrow) * K + lcol;
    const __bf16* gB = BT + (size_t)(n0 + lrow) * K + lcol;

    for (int k0 = 0; k0 < K; k0 += 32) {
        *(uint4*)(As + lrow * 32 + lcol)     = *(const uint4*)(gA + k0);
        *(uint4*)(As + lrow * 32 + lcol + 8) = *(const uint4*)(gA + k0 + 8);
        *(uint4*)(Bs + lrow * 32 + lcol)     = *(const uint4*)(gB + k0);
        *(uint4*)(Bs + lrow * 32 + lcol + 8) = *(const uint4*)(gB + k0 + 8);
        if (k0 + 32 < K) {                       // global_prefetch_b8 next tiles
            __builtin_prefetch(gA + k0 + 32, 0, 1);
            __builtin_prefetch(gB + k0 + 32, 0, 1);
        }
        __syncthreads();
        v16bf bfr[2], afr[4];
        for (int nt = 0; nt < 2; ++nt)
            bfr[nt] = ld_frag(Bs + (wn * 32 + nt * 16) * 32, 32);
        for (int mt = 0; mt < 4; ++mt)
            afr[mt] = ld_frag(As + (wm * 64 + mt * 16) * 32, 32);
        for (int mt = 0; mt < 4; ++mt)
            for (int nt = 0; nt < 2; ++nt)
                acc[mt * 2 + nt] = __builtin_amdgcn_wmma_f32_16x16x32_bf16(
                    false, afr[mt], false, bfr[nt], (short)0, acc[mt * 2 + nt], false, false);
        __syncthreads();
    }

    int rbase = m0 + wm * 64 + ((lane & 16) ? 8 : 0);
    for (int mt = 0; mt < 4; ++mt)
        for (int nt = 0; nt < 2; ++nt) {
            int col = n0 + wn * 32 + nt * 16 + (lane & 15);
            float bv = bias[col];
            for (int e = 0; e < 8; ++e) {
                int row = rbase + mt * 16 + e;
                float val = acc[mt * 2 + nt][e] + bv;
                if constexpr (BF16OUT)
                    ((__bf16*)Cout)[(size_t)row * N + col] = f2bf(val);
                else
                    ((float*)Cout)[(size_t)row * N + col] = val;
            }
        }
}

// ------------------------------------------------------------- flash attention
// 32 queries per workgroup, 64-key chunks, head dim 512, exact online softmax.
// 8 waves: (qm = w>>2) picks 16-query half, (dn = w&3) picks 128-wide D slice.
#define QB  32
#define KVB 64
#define HD  512

__global__ __launch_bounds__(256)
void flash_attn_kernel(const __bf16* __restrict__ Q, const __bf16* __restrict__ Km,
                       const __bf16* __restrict__ Vm, __bf16* __restrict__ Oa,
                       int S, float scale) {
    extern __shared__ __align__(16) char smem[];
    __bf16* Qs  = (__bf16*)smem;                                 // QB  x HD
    __bf16* Ks  = (__bf16*)(smem + (size_t)QB * HD * 2);         // KVB x HD
    __bf16* VT  = (__bf16*)(smem + (size_t)(QB + KVB) * HD * 2); // HD x KVB (V^T)
    float*  Sb  = (float*)(smem + (size_t)(QB + 2 * KVB) * HD * 2);  // QB x KVB
    __bf16* Pb  = (__bf16*)((char*)Sb + (size_t)QB * KVB * 4);       // QB x KVB
    float*  mrow = (float*)((char*)Pb + (size_t)QB * KVB * 2);
    float*  lrow = mrow + QB;
    float*  rrow = lrow + QB;
    float*  red  = rrow + QB;                                    // QB x 16 scratch

    int tid = threadIdx.x, lane = tid & 31, w = tid >> 5;
    int qm = w >> 2, dn = w & 3;
    int q0 = blockIdx.x * QB;
    size_t bbase = (size_t)blockIdx.y * S * HD;

#ifdef HAVE_TDM
    if (w == 0)    // Q tile via Tensor Data Mover (async DMA into LDS)
        tdm_load_2d((uint32_t)((const char*)Qs - smem),
                    Q + bbase + (size_t)q0 * HD, QB, HD, HD);
#else
    {   // stage Q tile manually
        int row = tid >> 3, c0 = (tid & 7) * 64;
        const uint4* src = (const uint4*)(Q + bbase + (size_t)(q0 + row) * HD + c0);
        uint4* dst = (uint4*)(Qs + row * HD + c0);
        for (int j = 0; j < 8; ++j) dst[j] = src[j];
    }
#endif
    if (tid < QB) { mrow[tid] = -__builtin_inff(); lrow[tid] = 0.0f; }

    v8f acc[8];
    for (int t = 0; t < 8; ++t)
        for (int e = 0; e < 8; ++e) acc[t][e] = 0.0f;

    for (int kv0 = 0; kv0 < S; kv0 += KVB) {
        __syncthreads();   // previous PV reads done before overwriting Ks/VT
#ifdef HAVE_TDM
        if (w == 0) {      // K chunk via TDM; covers the Q load too on iter 0
            tdm_load_2d((uint32_t)((const char*)Ks - smem),
                        Km + bbase + (size_t)kv0 * HD, KVB, HD, HD);
            __builtin_amdgcn_s_wait_tensorcnt(0);
        }
#else
        {   // stage K chunk (row-major: exactly the B^T layout scores need)
            int row = tid >> 2, c0 = (tid & 3) * 128;
            const uint4* src = (const uint4*)(Km + bbase + (size_t)(kv0 + row) * HD + c0);
            uint4* dst = (uint4*)(Ks + row * HD + c0);
            for (int j = 0; j < 16; ++j) dst[j] = src[j];
        }
#endif
        {   // stage V chunk transposed: VT[d][k] so PV B-frags read contiguous K
            int kk = tid >> 2, c0 = (tid & 3) * 128;
            const uint4* src = (const uint4*)(Vm + bbase + (size_t)(kv0 + kk) * HD + c0);
            for (int j = 0; j < 16; ++j) {
                Pack8 p; p.q = src[j];
                int d = c0 + j * 8;
                for (int e = 0; e < 8; ++e) VT[(d + e) * KVB + kk] = p.b[e];
            }
        }
        __syncthreads();

        {   // scores: wave (qm, kn) owns one 16x16 tile; contract over D=512
            int kn = dn;
            v8f s;
            for (int e = 0; e < 8; ++e) s[e] = 0.0f;
            for (int ks = 0; ks < HD; ks += 64) {   // 2 K-steps per trip, batched
                v16bf a0 = ld_frag(Qs + (qm * 16) * HD + ks, HD);
                v16bf b0 = ld_frag(Ks + (kn * 16) * HD + ks, HD);
                v16bf a1 = ld_frag(Qs + (qm * 16) * HD + ks + 32, HD);
                v16bf b1 = ld_frag(Ks + (kn * 16) * HD + ks + 32, HD);
                s = __builtin_amdgcn_wmma_f32_16x16x32_bf16(
                        false, a0, false, b0, (short)0, s, false, false);
                s = __builtin_amdgcn_wmma_f32_16x16x32_bf16(
                        false, a1, false, b1, (short)0, s, false, false);
            }
            int col = kn * 16 + (lane & 15);
            int rb  = qm * 16 + ((lane & 16) ? 8 : 0);
            for (int e = 0; e < 8; ++e) Sb[(rb + e) * KVB + col] = s[e] * scale;
        }
        __syncthreads();

        {   // online softmax: 8 threads per row (r = tid>>3, c = tid&7)
            int r = tid >> 3, c = tid & 7;
            float m_old = mrow[r];
            float pm = -__builtin_inff();
            for (int j = 0; j < 8; ++j)
                pm = fmaxf(pm, Sb[r * KVB + c * 8 + j]);
            red[r * 16 + c] = pm;
            __syncthreads();
            float mx = m_old;
            for (int t = 0; t < 8; ++t) mx = fmaxf(mx, red[r * 16 + t]);
            float sum = 0.0f;
            for (int j = 0; j < 8; ++j) {
                float p = __expf(Sb[r * KVB + c * 8 + j] - mx);
                sum += p;
                Pb[r * KVB + c * 8 + j] = f2bf(p);
            }
            red[r * 16 + 8 + c] = sum;
            __syncthreads();
            if (c == 0) {
                float s8 = 0.0f;
                for (int t = 0; t < 8; ++t) s8 += red[r * 16 + 8 + t];
                float rf = __expf(m_old - mx);
                mrow[r] = mx; lrow[r] = lrow[r] * rf + s8; rrow[r] = rf;
            }
        }
        __syncthreads();

        {   // rescale accumulators, then PV: acc += P(16x64) x V(64x128-slice)
            int rb = qm * 16 + ((lane & 16) ? 8 : 0);
            float rf[8];
            for (int e = 0; e < 8; ++e) rf[e] = rrow[rb + e];
            for (int t = 0; t < 8; ++t)
                for (int e = 0; e < 8; ++e) acc[t][e] *= rf[e];
            for (int kk = 0; kk < KVB; kk += 32) {
                v16bf a = ld_frag(Pb + (qm * 16) * KVB + kk, KVB);
                for (int th = 0; th < 8; th += 4) {     // batch 4 B-frags / burst
                    v16bf bb[4];
                    for (int j = 0; j < 4; ++j)
                        bb[j] = ld_frag(VT + (dn * 128 + (th + j) * 16) * KVB + kk, KVB);
                    for (int j = 0; j < 4; ++j)
                        acc[th + j] = __builtin_amdgcn_wmma_f32_16x16x32_bf16(
                            false, a, false, bb[j], (short)0, acc[th + j], false, false);
                }
            }
        }
    }
    __syncthreads();
    {   // finalize: divide by row sum, emit bf16 attn for the output GEMM
        int rb = qm * 16 + ((lane & 16) ? 8 : 0);
        for (int t = 0; t < 8; ++t) {
            int col = dn * 128 + t * 16 + (lane & 15);
            for (int e = 0; e < 8; ++e) {
                int row = rb + e;
                float val = acc[t][e] / lrow[row];
                Oa[bbase + (size_t)(q0 + row) * HD + col] = f2bf(val);
            }
        }
    }
}

// ------------------------------------------------------------------- driver
extern "C" void kernel_launch(void* const* d_in, const int* in_sizes, int n_in,
                              void* d_out, int out_size, void* d_ws, size_t ws_size,
                              hipStream_t stream) {
    (void)in_sizes; (void)n_in; (void)out_size; (void)ws_size;
    const float* x  = (const float*)d_in[0];
    const float* Wq = (const float*)d_in[1];
    const float* bq = (const float*)d_in[2];
    const float* Wk = (const float*)d_in[3];
    const float* bk = (const float*)d_in[4];
    const float* Wv = (const float*)d_in[5];
    const float* bv = (const float*)d_in[6];
    const float* Wo = (const float*)d_in[7];
    const float* bo = (const float*)d_in[8];

    const int B = 4, S = 4096, C = 512, Dd = 512;
    const size_t M = (size_t)B * S;                 // 16384 tokens

    // workspace layout (bf16 staging): ~82 MB
    char* ws = (char*)d_ws;
    __bf16* xb  = (__bf16*)(ws);                                  // M x C
    __bf16* WqT = (__bf16*)(ws + (size_t)16 * 1024 * 1024);       // D x C each
    __bf16* WkT = WqT + (size_t)512 * 512;
    __bf16* WvT = WkT + (size_t)512 * 512;
    __bf16* WoT = WvT + (size_t)512 * 512;
    __bf16* qb  = (__bf16*)(ws + (size_t)18 * 1024 * 1024);       // M x D
    __bf16* kb  = qb + M * Dd;
    __bf16* vb  = kb + M * Dd;
    __bf16* ab  = vb + M * Dd;

    // 1) convert x to bf16
    cvt_bf16_kernel<<<dim3((unsigned)((M * C / 8 + 255) / 256)), 256, 0, stream>>>(
        x, xb, (int)(M * C));
    // 2) convert + transpose weights so GEMM B-frags are contiguous LDS reads
    unsigned tb = (512 * 512 + 255) / 256;
    cvt_transpose_kernel<<<tb, 256, 0, stream>>>(Wq, WqT, C, Dd);
    cvt_transpose_kernel<<<tb, 256, 0, stream>>>(Wk, WkT, C, Dd);
    cvt_transpose_kernel<<<tb, 256, 0, stream>>>(Wv, WvT, C, Dd);
    cvt_transpose_kernel<<<tb, 256, 0, stream>>>(Wo, WoT, Dd, Dd);

    // 3) QKV projections (bf16 out, f32 accumulate)
    dim3 gg((unsigned)(M / 128), (unsigned)(Dd / 128));
    gemm_bias_kernel<true><<<gg, 256, 0, stream>>>(xb, WqT, bq, qb, (int)M, Dd, C);
    gemm_bias_kernel<true><<<gg, 256, 0, stream>>>(xb, WkT, bk, kb, (int)M, Dd, C);
    gemm_bias_kernel<true><<<gg, 256, 0, stream>>>(xb, WvT, bv, vb, (int)M, Dd, C);

    // 4) flash attention (dynamic LDS ~174 KB of the 320 KB WGP pool)
    size_t smem = (size_t)(QB + 2 * KVB) * HD * 2      // Qs + Ks + VT
                + (size_t)QB * KVB * 4                 // Sb
                + (size_t)QB * KVB * 2                 // Pb
                + (size_t)3 * QB * 4                   // m/l/rescale rows
                + (size_t)QB * 16 * 4;                 // softmax reduction scratch
    hipFuncSetAttribute((const void*)flash_attn_kernel,
                        hipFuncAttributeMaxDynamicSharedMemorySize, (int)smem);
    flash_attn_kernel<<<dim3(S / QB, B), 256, smem, stream>>>(
        qb, kb, vb, ab, S, 0.044194173824159216f /* 512^-0.5 */);

    // 5) output projection (f32 out)
    gemm_bias_kernel<false><<<gg, 256, 0, stream>>>(ab, WoT, bo, d_out, (int)M, Dd, Dd);
}
